// Pretrain_38439957299923
// MI455X (gfx1250) — matrix operand order repaired and checked by reading, not compiled, not run on hardware
//
#include <hip/hip_runtime.h>

typedef __attribute__((ext_vector_type(16))) _Float16 v16h;
typedef __attribute__((ext_vector_type(8)))  _Float16 v8h;
typedef __attribute__((ext_vector_type(8)))  float    v8f;
typedef unsigned short u16;

constexpr int PN  = 4096;   // nodes
constexpr int PDI = 128;    // input dim
constexpr int PDH = 256;    // hidden dim
constexpr int PK  = 17;     // top-(K+1)
constexpr float PTINV = 5.0f;   // 1/TEMP
constexpr float PEPS  = 1e-8f;

// ---------------------------------------------------------------------------
// WMMA GEMM: C[M,Ncols](f32) = A[M,K](f16 row-major) * Bt[Ncols,K]^T (f16)
// One wave per 16x16 C tile, K step 32, v_wmma_f32_16x16x32_f16.
// A frag: lane L (m=L&15): 8 halves at k0+((L>>4)*16), 8 halves at +8.
// B frag: lane L (n=L&15): 16 halves at Bt[n][k0+(L>>4)*16].
// ---------------------------------------------------------------------------
__global__ void k_gemm_nt(const u16* Aus, const u16* Bus, float* C,
                          int M, int Ncols, int K) {
  const _Float16* A  = reinterpret_cast<const _Float16*>(Aus);
  const _Float16* Bt = reinterpret_cast<const _Float16*>(Bus);
  const int lane  = threadIdx.x & 31;
  const int wpb   = blockDim.x >> 5;
  const int tn    = Ncols >> 4;
  const int total = (M >> 4) * tn;
  const int step  = gridDim.x * wpb;
  const int mlane = lane & 15;
  const int hi    = lane >> 4;

  for (int tile = blockIdx.x * wpb + (threadIdx.x >> 5); tile < total; tile += step) {
    const int tm = tile / tn;
    const int m0 = tm << 4;
    const int n0 = (tile - tm * tn) << 4;
    const _Float16* ap = A  + (size_t)(m0 + mlane) * K + hi * 16;
    const _Float16* bp = Bt + (size_t)(n0 + mlane) * K + hi * 16;
    v8f acc = {0.f, 0.f, 0.f, 0.f, 0.f, 0.f, 0.f, 0.f};
    for (int k0 = 0; k0 < K; k0 += 32) {
      v8h alo = *(const v8h*)(ap + k0);
      v8h ahi = *(const v8h*)(ap + k0 + 8);
      v16h a;
#pragma unroll
      for (int i = 0; i < 8; ++i) { a[i] = alo[i]; a[8 + i] = ahi[i]; }
      v16h b = *(const v16h*)(bp + k0);
      acc = __builtin_amdgcn_wmma_f32_16x16x32_f16(
          false, a, false, b, (short)0, acc, false, false);
    }
    float* cp = C + (size_t)(m0 + hi * 8) * Ncols + (n0 + mlane);
#pragma unroll
    for (int r = 0; r < 8; ++r) cp[(size_t)r * Ncols] = acc[r];
  }
}

// x1 = relu(x * p_feat) * p_shared  (f32 + f16 copies)
__global__ void k_x1(const float* x, const float* pf, const float* ps,
                     float* x1, u16* x1h, long total) {
  _Float16* xh = reinterpret_cast<_Float16*>(x1h);
  for (long t = blockIdx.x * (long)blockDim.x + threadIdx.x; t < total;
       t += (long)gridDim.x * blockDim.x) {
    int f = (int)(t & (long)(PDI - 1));
    float v = fmaxf(x[t] * pf[f], 0.f) * ps[f];
    x1[t] = v;
    xh[t] = (_Float16)v;
  }
}

__global__ void k_fill(float* p, long total, float v) {
  for (long t = blockIdx.x * (long)blockDim.x + threadIdx.x; t < total;
       t += (long)gridDim.x * blockDim.x) p[t] = v;
}

__global__ void k_deg(const int* dst, float* deg, int E) {
  int t = blockIdx.x * blockDim.x + threadIdx.x;
  if (t < E) atomicAdd(&deg[dst[t]], 1.0f);
}

__global__ void k_dinv(const float* deg, float* dinv, int n) {
  int t = blockIdx.x * blockDim.x + threadIdx.x;
  if (t < n) dinv[t] = (deg[t] > 0.f) ? rsqrtf(deg[t]) : 0.f;
}

// out = dinv[i]^2 * in   (self-loop term initializes the propagate output)
__global__ void k_prop_self(const float* in, const float* dinv, float* out,
                            int F, long total) {
  for (long t = blockIdx.x * (long)blockDim.x + threadIdx.x; t < total;
       t += (long)gridDim.x * blockDim.x) {
    int i = (int)(t / (long)F);
    float dv = dinv[i];
    out[t] = dv * dv * in[t];
  }
}

// out[dst] += dinv[s]*dinv[d]*in[src]   (thread per edge*feature)
__global__ void k_prop_edges(const int* src, const int* dst, const float* dinv,
                             const float* in, float* out, int E, int F) {
  long total = (long)E * (long)F;
  for (long t = blockIdx.x * (long)blockDim.x + threadIdx.x; t < total;
       t += (long)gridDim.x * blockDim.x) {
    int e = (int)(t / (long)F);
    int f = (int)(t - (long)e * (long)F);
    int s = src[e];
    int d = dst[e];
    float w = dinv[s] * dinv[d];
    atomicAdd(&out[(long)d * F + f], w * in[(long)s * F + f]);
  }
}

// h = concat([x1, agg]) * p_balance; hn = h/(||h||+eps) -> f16 (block 256/row)
__global__ void k_hn(const float* x1, const float* agg, const float* pbal, u16* hnh) {
  _Float16* hn = reinterpret_cast<_Float16*>(hnh);
  int i = blockIdx.x;
  int t = threadIdx.x;
  float v;
  if (t < PDI) v = x1[(long)i * PDI + t] * pbal[t];
  else         v = agg[(long)i * PDI + (t - PDI)] * pbal[t];
  __shared__ float s[PDH];
  s[t] = v * v;
  __syncthreads();
  for (int r = PDH >> 1; r > 0; r >>= 1) {
    if (t < r) s[t] += s[t + r];
    __syncthreads();
  }
  float inv = 1.0f / (sqrtf(s[0]) + PEPS);
  hn[(long)i * PDH + t] = (_Float16)(v * inv);
}

// f32 [R,C] -> f16 [C,R]
__global__ void k_tcast(const float* in, u16* outh, int R, int C) {
  _Float16* out = reinterpret_cast<_Float16*>(outh);
  long total = (long)R * (long)C;
  for (long t = blockIdx.x * (long)blockDim.x + threadIdx.x; t < total;
       t += (long)gridDim.x * blockDim.x) {
    int r = (int)(t / (long)C);
    int c = (int)(t - (long)r * (long)C);
    out[(long)c * R + r] = (_Float16)in[t];
  }
}

// top-(K+1) of each 4096-wide row; selected entries are knocked to -inf
__global__ void k_topk(const float* sim, float* vals, int* idxs) {
  int row = blockIdx.x;
  int t = threadIdx.x; // 256 threads, 16 elems each
  const float* r = sim + (long)row * PN;
  float lv[16];
#pragma unroll
  for (int i = 0; i < 16; ++i) lv[i] = r[i * 256 + t];
  __shared__ float sv[256];
  __shared__ int   si[256];
  for (int k = 0; k < PK; ++k) {
    float best = -3.0e38f;
    int bi = 0;
#pragma unroll
    for (int i = 0; i < 16; ++i) {
      if (lv[i] > best) { best = lv[i]; bi = i; }
    }
    sv[t] = best;
    si[t] = bi * 256 + t;
    __syncthreads();
    for (int s2 = 128; s2 > 0; s2 >>= 1) {
      if (t < s2 && sv[t + s2] > sv[t]) { sv[t] = sv[t + s2]; si[t] = si[t + s2]; }
      __syncthreads();
    }
    int bidx = si[0];
    float bval = sv[0];
    int slot = bidx >> 8;
    bool own = (t == (bidx & 255));
#pragma unroll
    for (int i = 0; i < 16; ++i) {
      if (own && i == slot) lv[i] = -3.0e38f;
    }
    if (t == 0) {
      if (bval != bval) bval = 0.f; // nan_to_num
      vals[(long)row * PK + k] = bval;
      idxs[(long)row * PK + k] = bidx;
    }
    __syncthreads();
  }
}

// Wund[i][j] = relu((a_ij + a_ji)/max(cnt,1)) via the two top-k lists
__global__ void k_wund(const float* vals, const int* idxs, float* Wf, u16* Whh) {
  _Float16* Wh = reinterpret_cast<_Float16*>(Whh);
  long total = (long)PN * (long)PN;
  for (long t = blockIdx.x * (long)blockDim.x + threadIdx.x; t < total;
       t += (long)gridDim.x * blockDim.x) {
    int i = (int)(t / (long)PN);
    int j = (int)(t - (long)i * (long)PN);
    float a = 0.f;
    int c = 0;
#pragma unroll
    for (int k = 0; k < PK; ++k) {
      if (idxs[(long)i * PK + k] == j) { a += vals[(long)i * PK + k]; ++c; }
      if (idxs[(long)j * PK + k] == i) { a += vals[(long)j * PK + k]; ++c; }
    }
    float w = 0.f;
    if (c > 0) { w = a / (float)c; w = fmaxf(w, 0.f); }
    Wf[t] = w;
    Wh[t] = (_Float16)w;
  }
}

// x = relu(x + bias[col]); also writes f16 copy
__global__ void k_bias_relu(float* x, const float* bias, u16* outh, int F, long total) {
  _Float16* o = reinterpret_cast<_Float16*>(outh);
  for (long t = blockIdx.x * (long)blockDim.x + threadIdx.x; t < total;
       t += (long)gridDim.x * blockDim.x) {
    int c = (int)(t % (long)F);
    float v = fmaxf(x[t] + bias[c], 0.f);
    x[t] = v;
    o[t] = (_Float16)v;
  }
}

// x = elu(x + bias[col])
__global__ void k_bias_elu(float* x, const float* bias, int F, long total) {
  for (long t = blockIdx.x * (long)blockDim.x + threadIdx.x; t < total;
       t += (long)gridDim.x * blockDim.x) {
    int c = (int)(t % (long)F);
    float v = x[t] + bias[c];
    x[t] = (v > 0.f) ? v : (expf(v) - 1.f);
  }
}

// row L2-normalize (256 wide) -> f16
__global__ void k_rownorm16(const float* in, u16* outh) {
  _Float16* out = reinterpret_cast<_Float16*>(outh);
  int i = blockIdx.x;
  int t = threadIdx.x;
  float v = in[(long)i * PDH + t];
  __shared__ float s[PDH];
  s[t] = v * v;
  __syncthreads();
  for (int r = PDH >> 1; r > 0; r >>= 1) {
    if (t < r) s[t] += s[t + r];
    __syncthreads();
  }
  float inv = 1.0f / (sqrtf(s[0]) + PEPS);
  out[(long)i * PDH + t] = (_Float16)(v * inv);
}

// in-place row softmax of P/TEMP (4096 wide; 256 threads x 16)
__global__ void k_softmax(float* P) {
  int row = blockIdx.x;
  int t = threadIdx.x;
  float* r = P + (long)row * PN;
  float lv[16];
  float mx = -3.0e38f;
#pragma unroll
  for (int i = 0; i < 16; ++i) {
    lv[i] = r[i * 256 + t] * PTINV;
    mx = fmaxf(mx, lv[i]);
  }
  __shared__ float s[256];
  s[t] = mx;
  __syncthreads();
  for (int r2 = 128; r2 > 0; r2 >>= 1) {
    if (t < r2) s[t] = fmaxf(s[t], s[t + r2]);
    __syncthreads();
  }
  float m = s[0];
  __syncthreads();
  float sum = 0.f;
#pragma unroll
  for (int i = 0; i < 16; ++i) { lv[i] = expf(lv[i] - m); sum += lv[i]; }
  s[t] = sum;
  __syncthreads();
  for (int r2 = 128; r2 > 0; r2 >>= 1) {
    if (t < r2) s[t] += s[t + r2];
    __syncthreads();
  }
  float inv = 1.0f / s[0];
#pragma unroll
  for (int i = 0; i < 16; ++i) r[i * 256 + t] = lv[i] * inv;
}

__global__ void k_zero_acc(float* acc) {
  if (threadIdx.x == 0) { acc[0] = 0.f; acc[1] = 0.f; acc[2] = 0.f; }
}

__global__ void k_loss_self(const float* P, float* acc) {
  int i = blockIdx.x * blockDim.x + threadIdx.x;
  if (i < PN) atomicAdd(&acc[0], -logf(fmaxf(P[(long)i * PN + i], PEPS)));
}

// which==0: pf[i]=sum_j P[i][j]*W[i][j]; which==1: pb[i]=sum_j P[j][i]*W[i][j]
__global__ void k_loss_row(const float* P, const float* W, float* acc, int which) {
  int row = blockIdx.x;
  int t = threadIdx.x;
  float sum = 0.f;
  if (which == 0) {
#pragma unroll
    for (int i = 0; i < 16; ++i) {
      int j = i * 256 + t;
      sum += P[(long)row * PN + j] * W[(long)row * PN + j];
    }
  } else {
#pragma unroll
    for (int i = 0; i < 16; ++i) {
      int j = i * 256 + t;
      sum += P[(long)j * PN + row] * W[(long)row * PN + j];
    }
  }
  __shared__ float s[256];
  s[t] = sum;
  __syncthreads();
  for (int r2 = 128; r2 > 0; r2 >>= 1) {
    if (t < r2) s[t] += s[t + r2];
    __syncthreads();
  }
  if (t == 0) atomicAdd(&acc[1 + which], -logf(fmaxf(s[0], PEPS)));
}

__global__ void k_final(const float* acc, float* out) {
  if (threadIdx.x == 0) {
    float invn = 1.0f / (float)PN;
    out[0] = acc[0] * invn + 0.5f * (acc[1] * invn + acc[2] * invn);
  }
}

// ---------------------------------------------------------------------------
extern "C" void kernel_launch(void* const* d_in, const int* in_sizes, int n_in,
                              void* d_out, int out_size, void* d_ws, size_t ws_size,
                              hipStream_t stream) {
  const float* x        = (const float*)d_in[0];
  const int*   esrc     = (const int*)d_in[1];
  const int*   edst     = (const int*)d_in[2];
  const float* p_feat   = (const float*)d_in[3];
  const float* p_shared = (const float*)d_in[4];
  const float* p_bal    = (const float*)d_in[5];
  const float* W1       = (const float*)d_in[6];
  const float* b1       = (const float*)d_in[7];
  const float* W2       = (const float*)d_in[8];
  const float* b2       = (const float*)d_in[9];
  float* out = (float*)d_out;
  const int E = in_sizes[1];

  // workspace carve (64B-aligned slabs)
  char* wsp = (char*)d_ws;
  size_t off = 0;
  float* simP  = (float*)(wsp + off); off += (size_t)PN * PN * 4;          // sim / prob
  float* WundF = (float*)(wsp + off); off += (size_t)PN * PN * 4;
  u16*   WundH = (u16*)(wsp + off);   off += (size_t)PN * PN * 2;
  float* x1F   = (float*)(wsp + off); off += (size_t)PN * PDI * 4;
  u16*   x1H   = (u16*)(wsp + off);   off += (size_t)PN * PDI * 2;
  float* deg   = (float*)(wsp + off); off += (size_t)PN * 4;
  float* dinv  = (float*)(wsp + off); off += (size_t)PN * 4;
  float* aggF  = (float*)(wsp + off); off += (size_t)PN * PDI * 4;
  u16*   hnH   = (u16*)(wsp + off);   off += (size_t)PN * PDH * 2;
  float* vals  = (float*)(wsp + off); off += (size_t)PN * PK * 4 + 192;
  int*   idxs  = (int*)(wsp + off);   off += (size_t)PN * PK * 4 + 192;
  u16*   W1tH  = (u16*)(wsp + off);   off += (size_t)PDH * PDI * 2;
  u16*   W2tH  = (u16*)(wsp + off);   off += (size_t)PDH * PDH * 2;
  float* XW1F  = (float*)(wsp + off); off += (size_t)PN * PDH * 4;
  u16*   XW1tH = (u16*)(wsp + off);   off += (size_t)PN * PDH * 2;         // [DH,N]
  float* H1F   = (float*)(wsp + off); off += (size_t)PN * PDH * 4;
  u16*   h1H   = (u16*)(wsp + off);   off += (size_t)PN * PDH * 2;
  float* HW2F  = (float*)(wsp + off); off += (size_t)PN * PDH * 4;
  float* Z1F   = (float*)(wsp + off); off += (size_t)PN * PDH * 4;
  u16*   z1nH  = (u16*)(wsp + off);   off += (size_t)PN * PDH * 2;
  float* G1F   = (float*)(wsp + off); off += (size_t)PN * PDH * 4;
  u16*   g1H   = (u16*)(wsp + off);   off += (size_t)PN * PDH * 2;
  float* GW2F  = (float*)(wsp + off); off += (size_t)PN * PDH * 4;
  u16*   GW2tH = (u16*)(wsp + off);   off += (size_t)PN * PDH * 2;         // [DH,N]
  float* Z2F   = (float*)(wsp + off); off += (size_t)PN * PDH * 4;
  u16*   z2nH  = (u16*)(wsp + off);   off += (size_t)PN * PDH * 2;
  float* acc   = (float*)(wsp + off); off += 256;

  long tNDI = (long)PN * PDI;
  long tNDH = (long)PN * PDH;
  long tNN  = (long)PN * PN;
  unsigned bNDI = (unsigned)((tNDI + 255) / 256);
  unsigned bNDH = (unsigned)((tNDH + 255) / 256);
  unsigned bNN  = (unsigned)((tNN + 255) / 256);
  unsigned bEDI = (unsigned)(((long)E * PDI + 255) / 256);
  unsigned bEDH = (unsigned)(((long)E * PDH + 255) / 256);
  int tilesNN = (PN >> 4) * (PN >> 4);
  int tilesND = (PN >> 4) * (PDH >> 4);

  // 1) prompts
  k_x1<<<bNDI, 256, 0, stream>>>(x, p_feat, p_shared, x1F, x1H, tNDI);
  // 2) degrees (self loop = 1) -> dinv
  k_fill<<<16, 256, 0, stream>>>(deg, (long)PN, 1.0f);
  k_deg<<<(E + 255) / 256, 256, 0, stream>>>(edst, deg, E);
  k_dinv<<<(PN + 255) / 256, 256, 0, stream>>>(deg, dinv, PN);
  // 3) agg = A_norm @ x1
  k_prop_self<<<bNDI, 256, 0, stream>>>(x1F, dinv, aggF, PDI, tNDI);
  k_prop_edges<<<bEDI, 256, 0, stream>>>(esrc, edst, dinv, x1F, aggF, E, PDI);
  // 4) hn (f16)
  k_hn<<<PN, PDH, 0, stream>>>(x1F, aggF, p_bal, hnH);
  // 5) weight transposes
  k_tcast<<<(PDI * PDH + 255) / 256, 256, 0, stream>>>(W1, W1tH, PDI, PDH);
  k_tcast<<<(PDH * PDH + 255) / 256, 256, 0, stream>>>(W2, W2tH, PDH, PDH);
  // 6) sim = hn @ hn^T
  k_gemm_nt<<<(tilesNN + 3) / 4, 128, 0, stream>>>(hnH, hnH, simP, PN, PN, PDH);
  // 7) top-(K+1)
  k_topk<<<PN, 256, 0, stream>>>(simP, vals, idxs);
  // 8) Wund
  k_wund<<<bNN, 256, 0, stream>>>(vals, idxs, WundF, WundH);
  // 9) XW1 = x1 @ W1 (+ transposed f16 copy)
  k_gemm_nt<<<(tilesND + 3) / 4, 128, 0, stream>>>(x1H, W1tH, XW1F, PN, PDH, PDI);
  k_tcast<<<bNDH, 256, 0, stream>>>(XW1F, XW1tH, PN, PDH);
  // 10) view 1: h1 = relu(prop(XW1)+b1); z1 = elu(prop(h1@W2)+b2)
  k_prop_self<<<bNDH, 256, 0, stream>>>(XW1F, dinv, H1F, PDH, tNDH);
  k_prop_edges<<<bEDH, 256, 0, stream>>>(esrc, edst, dinv, XW1F, H1F, E, PDH);
  k_bias_relu<<<bNDH, 256, 0, stream>>>(H1F, b1, h1H, PDH, tNDH);
  k_gemm_nt<<<(tilesND + 3) / 4, 128, 0, stream>>>(h1H, W2tH, HW2F, PN, PDH, PDH);
  k_prop_self<<<bNDH, 256, 0, stream>>>(HW2F, dinv, Z1F, PDH, tNDH);
  k_prop_edges<<<bEDH, 256, 0, stream>>>(esrc, edst, dinv, HW2F, Z1F, E, PDH);
  k_bias_elu<<<bNDH, 256, 0, stream>>>(Z1F, b2, PDH, tNDH);
  k_rownorm16<<<PN, PDH, 0, stream>>>(Z1F, z1nH);
  // 11) view 2: g1 = relu(Wund@XW1+b1); z2 = elu(Wund@(g1@W2)+b2)
  k_gemm_nt<<<(tilesND + 3) / 4, 128, 0, stream>>>(WundH, XW1tH, G1F, PN, PDH, PN);
  k_bias_relu<<<bNDH, 256, 0, stream>>>(G1F, b1, g1H, PDH, tNDH);
  k_gemm_nt<<<(tilesND + 3) / 4, 128, 0, stream>>>(g1H, W2tH, GW2F, PN, PDH, PDH);
  k_tcast<<<bNDH, 256, 0, stream>>>(GW2F, GW2tH, PN, PDH);
  k_gemm_nt<<<(tilesND + 3) / 4, 128, 0, stream>>>(WundH, GW2tH, Z2F, PN, PDH, PN);
  k_bias_elu<<<bNDH, 256, 0, stream>>>(Z2F, b2, PDH, tNDH);
  k_rownorm16<<<PN, PDH, 0, stream>>>(Z2F, z2nH);
  // 12) prob = softmax(z1n @ z2n^T / TEMP)
  k_gemm_nt<<<(tilesNN + 3) / 4, 128, 0, stream>>>(z1nH, z2nH, simP, PN, PN, PDH);
  k_softmax<<<PN, 256, 0, stream>>>(simP);
  // 13) losses
  k_zero_acc<<<1, 32, 0, stream>>>(acc);
  k_loss_self<<<(PN + 255) / 256, 256, 0, stream>>>(simP, acc);
  k_loss_row<<<PN, 256, 0, stream>>>(simP, WundF, acc, 0);
  k_loss_row<<<PN, 256, 0, stream>>>(simP, WundF, acc, 1);
  k_final<<<1, 32, 0, stream>>>(acc, out);

  (void)n_in; (void)out_size; (void)ws_size; (void)off;
}
